// EdgeUpdate_33337536151683
// MI455X (gfx1250) — compile-verified, matrix operand order checked by reading, not run on hardware
//
#include <hip/hip_runtime.h>
#include <hip/hip_bf16.h>

typedef __attribute__((ext_vector_type(16))) __bf16         v16bf;
typedef __attribute__((ext_vector_type(8)))  float          v8f;
typedef __attribute__((ext_vector_type(8)))  unsigned short ushort8;

// exact operand types for the async-LDS builtin (from hipcc diagnostic):
typedef int vsi4 __attribute__((vector_size(4 * sizeof(int))));
typedef __attribute__((address_space(1))) vsi4* g_vsi4_ptr;
typedef __attribute__((address_space(3))) vsi4* l_vsi4_ptr;

#if __has_builtin(__builtin_amdgcn_global_load_async_to_lds_b128)
#define USE_ASYNC_LDS 1
#else
#define USE_ASYNC_LDS 0
#endif

__device__ __forceinline__ unsigned short f2bf(float f) {
    unsigned int u = __float_as_uint(f);
    unsigned int r = (u + 0x7FFFu + ((u >> 16) & 1u)) >> 16;  // RNE
    return (unsigned short)r;
}

__device__ __forceinline__ void wait_async0() {
#if USE_ASYNC_LDS
#if __has_builtin(__builtin_amdgcn_s_wait_asynccnt)
    __builtin_amdgcn_s_wait_asynccnt(0);
#else
    asm volatile("s_wait_asynccnt 0x0" ::: "memory");
#endif
#endif
}

// ---------------------------------------------------------------------------
// Kernel 1: node LayerNorm, output bf16 rows (one wave32 per 128-wide row)
// ---------------------------------------------------------------------------
__global__ __launch_bounds__(256) void ln_pack_nodes(
    const float* __restrict__ x, const float* __restrict__ w,
    const float* __restrict__ b, unsigned short* __restrict__ xnb, int N) {
    int row  = (int)((blockIdx.x * blockDim.x + threadIdx.x) >> 5);
    int lane = threadIdx.x & 31;
    if (row >= N) return;
    const float4 v = *(const float4*)(x + (size_t)row * 128 + lane * 4);
    float s  = v.x + v.y + v.z + v.w;
    float ss = v.x * v.x + v.y * v.y + v.z * v.z + v.w * v.w;
    #pragma unroll
    for (int m = 1; m < 32; m <<= 1) {
        s  += __shfl_xor(s,  m, 32);
        ss += __shfl_xor(ss, m, 32);
    }
    float mu   = s * (1.0f / 128.0f);
    float var  = ss * (1.0f / 128.0f) - mu * mu;
    float rstd = rsqrtf(var + 1e-5f);
    const float4 wv = *(const float4*)(w + lane * 4);
    const float4 bv = *(const float4*)(b + lane * 4);
    unsigned int lo = (unsigned int)f2bf((v.x - mu) * rstd * wv.x + bv.x)
                    | ((unsigned int)f2bf((v.y - mu) * rstd * wv.y + bv.y) << 16);
    unsigned int hi = (unsigned int)f2bf((v.z - mu) * rstd * wv.z + bv.z)
                    | ((unsigned int)f2bf((v.w - mu) * rstd * wv.w + bv.w) << 16);
    uint2 o; o.x = lo; o.y = hi;
    *(uint2*)(xnb + (size_t)row * 128 + lane * 4) = o;
}

// ---------------------------------------------------------------------------
// Kernel 2: repack w1 (384x256 fp32) into bf16 B-operand order:
//   w1p[kt][nt][lane][16 halfs], lane holds column n = nt*16 + (lane&15),
//   lanes 0-15 hold K {kt*32 + 0..7, +16..23}, lanes 16-31 hold {+8..15, +24..31}
// ---------------------------------------------------------------------------
__global__ __launch_bounds__(256) void pack_w1(
    const float* __restrict__ w1, unsigned short* __restrict__ w1p) {
    int tid  = blockIdx.x * blockDim.x + threadIdx.x;   // 12*16*32*16 = 98304
    int h    = tid & 15;
    int lane = (tid >> 4) & 31;
    int nt   = (tid >> 9) & 15;
    int kt   = tid >> 13;
    int grp  = lane >> 4;
    int kl   = (h < 8 ? h : h + 8) + grp * 8;
    int k    = kt * 32 + kl;
    int n    = nt * 16 + (lane & 15);
    w1p[tid] = f2bf(w1[(size_t)k * 256 + n]);
}

// ---------------------------------------------------------------------------
// Kernel 3: edge GEMM + gate + residual + edge LayerNorm.
// Block = 256 threads = 8 waves; wave handles 16 edges x 256 outputs.
// B (w1 bf16) staged per-block into LDS, double-buffered via async loads.
// ---------------------------------------------------------------------------
union ABu { v16bf v; ushort8 h[2]; };

__device__ __forceinline__ ushort8 load_m8(
    const unsigned short* __restrict__ xnb, const float* __restrict__ eattr,
    long long src, long long dst, long long e, int col) {
    if (col < 128) return *(const ushort8*)(xnb + src * 128 + col);
    if (col < 256) return *(const ushort8*)(xnb + dst * 128 + (col - 128));
    const float4* p = (const float4*)(eattr + e * 128 + (col - 256));
    float4 a = p[0], b = p[1];
    ushort8 r;
    r[0] = f2bf(a.x); r[1] = f2bf(a.y); r[2] = f2bf(a.z); r[3] = f2bf(a.w);
    r[4] = f2bf(b.x); r[5] = f2bf(b.y); r[6] = f2bf(b.z); r[7] = f2bf(b.w);
    return r;
}

#define KT_TILE_USHORTS 8192   // 16 nt * 32 lanes * 16 halfs = 16KB per K-chunk

__global__ __launch_bounds__(256) void edge_gemm(
    const unsigned short* __restrict__ xnb, const float* __restrict__ eattr,
    const long long* __restrict__ ei, unsigned short* __restrict__ w1p,
    const float* __restrict__ b1, const float* __restrict__ enw,
    const float* __restrict__ enb, float* __restrict__ out, long long E) {
    __shared__ unsigned short Bsm[2][KT_TILE_USHORTS];   // 2 x 16KB double buffer

    const int t    = threadIdx.x;
    const int lane = t & 31;
    const int wave = t >> 5;
    const long long e0 = (long long)blockIdx.x * 128 + (long long)wave * 16;
    const bool active  = (e0 < E);          // wave-uniform (EXEC stays all-1s)
    const int grp = lane >> 4;              // which 16-lane half
    const int r   = lane & 15;              // A row / output column within tile

    long long ea = 0, src = 0, dst = 0;
    if (active) {
        ea  = e0 + r;
        src = ei[ea];
        dst = ei[E + ea];
    }

    // ---- stage B chunk kt into LDS buffer b (all 256 threads) -------------
    auto stage = [&](int kt, int b) {
        #pragma unroll
        for (int c = 0; c < 4; ++c) {
            const int chunk = t + c * 256;  // 0..1023, 16B each
#if USE_ASYNC_LDS
            __builtin_amdgcn_global_load_async_to_lds_b128(
                (g_vsi4_ptr)(w1p + (size_t)kt * KT_TILE_USHORTS + chunk * 8),
                (l_vsi4_ptr)(&Bsm[b][chunk * 8]),
                0, 0);
#else
            *(ushort8*)(&Bsm[b][chunk * 8]) =
                *(const ushort8*)(w1p + (size_t)kt * KT_TILE_USHORTS + chunk * 8);
#endif
        }
    };

    v8f acc[16];
    #pragma unroll
    for (int i = 0; i < 16; ++i)
        #pragma unroll
        for (int j = 0; j < 8; ++j) acc[i][j] = 0.0f;

    // prologue: stage first K-chunk
    stage(0, 0);
    wait_async0();
    __syncthreads();

    // ---- GEMM: K = 384 in 12 chunks of 32, N = 256 in 16 tiles of 16 ----
    #pragma unroll
    for (int kt = 0; kt < 12; ++kt) {
        const int cur = kt & 1;
        if (kt + 1 < 12) stage(kt + 1, cur ^ 1);   // overlap next fill

        if (active) {
            ABu a;
            const int kA = kt * 32 + grp * 8;
            a.h[0] = load_m8(xnb, eattr, src, dst, ea, kA);
            a.h[1] = load_m8(xnb, eattr, src, dst, ea, kA + 16);
            #pragma unroll
            for (int nt = 0; nt < 16; ++nt) {
                ABu bm;
                const unsigned short* q = &Bsm[cur][(nt * 32 + lane) * 16];
                bm.h[0] = *(const ushort8*)(q);
                bm.h[1] = *(const ushort8*)(q + 8);
                acc[nt] = __builtin_amdgcn_wmma_f32_16x16x32_bf16(
                    false, a.v, false, bm.v, (short)0, acc[nt], false, false);
            }
        }

        wait_async0();        // next chunk landed in LDS
        __syncthreads();      // everyone done reading cur before it is refilled
    }

    if (!active) return;

    // ---- epilogue: bias, relu, sigmoid gate, residual, edge LayerNorm ----
    float bias[16], ew8[8], eb8[8];
    #pragma unroll
    for (int nt = 0; nt < 16; ++nt) bias[nt] = b1[nt * 16 + r];
    #pragma unroll
    for (int nt = 0; nt < 8; ++nt) {
        int j = nt * 16 + r;
        ew8[nt] = enw[j];
        eb8[nt] = enb[j];
    }

    // C layout: lane holds column n = nt*16 + r; acc[nt][v] is row v + grp*8
    #pragma unroll
    for (int v = 0; v < 8; ++v) {
        const long long erow = e0 + v + grp * 8;
        float y[8], s = 0.0f, ssq = 0.0f;
        #pragma unroll
        for (int nt = 0; nt < 8; ++nt) {
            const int j = nt * 16 + r;
            float d = acc[nt][v] + bias[nt];
            d = d > 0.0f ? d : 0.0f;                       // relu(delta)
            float g = acc[nt + 8][v] + bias[nt + 8];
            g = g > 0.0f ? g : 0.0f;                       // relu(gate logit)
            g = 1.0f / (1.0f + __expf(-g));                // sigmoid
            float yy = eattr[erow * 128 + j] + g * d;      // residual
            y[nt] = yy;
            s += yy;
            ssq += yy * yy;
        }
        // row reduction across the 16 lanes of this half-wave
        #pragma unroll
        for (int m = 1; m < 16; m <<= 1) {
            s   += __shfl_xor(s,   m, 32);
            ssq += __shfl_xor(ssq, m, 32);
        }
        const float mu   = s * (1.0f / 128.0f);
        const float var  = ssq * (1.0f / 128.0f) - mu * mu;
        const float rstd = rsqrtf(var + 1e-5f);
        #pragma unroll
        for (int nt = 0; nt < 8; ++nt) {
            const int j = nt * 16 + r;
            out[erow * 128 + j] = (y[nt] - mu) * rstd * ew8[nt] + eb8[nt];
        }
    }
}

// ---------------------------------------------------------------------------
extern "C" void kernel_launch(void* const* d_in, const int* in_sizes, int n_in,
                              void* d_out, int out_size, void* d_ws, size_t ws_size,
                              hipStream_t stream) {
    const float*     x     = (const float*)d_in[0];
    const long long* ei    = (const long long*)d_in[1];
    const float*     eattr = (const float*)d_in[2];
    const float*     nnw   = (const float*)d_in[3];
    const float*     nnb   = (const float*)d_in[4];
    const float*     w1    = (const float*)d_in[5];
    const float*     b1    = (const float*)d_in[6];
    const float*     enw   = (const float*)d_in[7];
    const float*     enb   = (const float*)d_in[8];
    float*           out   = (float*)d_out;

    const int       N = in_sizes[0] / 128;
    const long long E = (long long)(in_sizes[2] / 128);

    unsigned short* xnb = (unsigned short*)d_ws;            // N*128 bf16
    unsigned short* w1p = xnb + (size_t)N * 128;            // 12*16*32*16 bf16

    // 1) node LayerNorm -> bf16 (wave per row)
    {
        int blocks = (N + 7) / 8;
        ln_pack_nodes<<<blocks, 256, 0, stream>>>(x, nnw, nnb, xnb, N);
    }
    // 2) repack w1 into WMMA B-operand order (bf16)
    {
        int total = 12 * 16 * 32 * 16;   // 98304
        pack_w1<<<total / 256, 256, 0, stream>>>(w1, w1p);
    }
    // 3) fused edge GEMM + gate + residual + LayerNorm (LDS-staged B)
    {
        int blocks = (int)((E + 127) / 128);
        edge_gemm<<<blocks, 256, 0, stream>>>(xnb, eattr, ei, w1p, b1, enw,
                                              enb, out, E);
    }
}
